// MLP_SparseLinear_Monarch_Deform_27504970564322
// MI455X (gfx1250) — matrix operand order repaired and checked by reading, not compiled, not run on hardware
//
#include <hip/hip_runtime.h>
#include <hip/hip_bf16.h>
#include <stdint.h>

// ---------------------------------------------------------------------------
// Fused Monarch block-MLP for MI455X (gfx1250, wave32, WMMA).
//
// Roofline: ~276 MB mandatory HBM traffic -> ~11.9us @ 23.3 TB/s; 34 GFLOP of
// bf16 WMMA math is negligible. So: ONE fused kernel, intermediates in LDS.
//   Per WG (8 batch rows, 4 waves / 128 thr):
//     phase A: y0[s][b][k0]  = x-block @ w0          (64 blocks)
//     phase B: z[t][b][s]    = elu(y0-col @ w1 + b1) @ w2   (128 s)
//     phase C: out           = z-col @ w3 + b3       (64 t)
// Weights are pre-repacked (prologue kernel) to bf16 WMMA lane-order tiles in
// d_ws (4 MB); staging into LDS is global_load_async_to_lds_b128 tracked by
// ASYNCcnt, double-buffered so the next tile streams while WMMA runs.
// A rows 8..15 are junk by construction (only D rows 0..7 are stored), so
// operand loads are unpredicated -- no exec-mask churn in the inner loops.
// bf16 conversion uses native fptrunc (RNE) so the backend can pick
// v_cvt_pk_bf16_f32 instead of a manual bit-twiddle sequence.
// ---------------------------------------------------------------------------

typedef __attribute__((ext_vector_type(16))) __bf16 v16bf;
typedef __attribute__((ext_vector_type(2)))  __bf16 v2bf;
typedef __attribute__((ext_vector_type(8)))  float  v8f;
typedef unsigned int  u32;
typedef unsigned short u16;
typedef __attribute__((ext_vector_type(4))) u32 u32x4;
typedef __attribute__((__vector_size__(4 * sizeof(int)))) int v4i;  // builtin's pointee type

#define NROWS    8192
#define IN_DIM   4096
#define MT       8          // batch rows per workgroup
#define NTHREADS 128        // 4 waves
// ws layout (bf16 elements): lane-order 32x16 tiles, 512 elems each
#define W0_OFF   0              // [n=64][kc=2][nt=8][512]
#define W1_OFF   (512*1024)     // [s=128][kc=2][nt=4][512]
#define W2_OFF   (1024*1024)    // [s=128][kc=2][nt=4][512]
#define W3_OFF   (1536*1024)    // [t=64][kc=4][nt=4][512]
#define WS_ELEMS (2048*1024)    // 2M bf16 = 4 MB

// Native float->bf16 (fptrunc = round-to-nearest-even, matches jnp/torch).
static __device__ __forceinline__ u16 f2bf(float f) {
  return __builtin_bit_cast(u16, (__bf16)f);
}
static __device__ __forceinline__ u32 f2bf2(float lo, float hi) {
  v2bf h; h.x = (__bf16)lo; h.y = (__bf16)hi;   // backend may fuse -> v_cvt_pk_bf16_f32
  return __builtin_bit_cast(u32, h);
}

struct Pair16 { u32x4 lo, hi; };

static __device__ __forceinline__ v16bf pack_ab(u32x4 lo, u32x4 hi) {
  Pair16 p; p.lo = lo; p.hi = hi;
  return __builtin_bit_cast(v16bf, p);
}

// A operand (16x32 bf16, M=batch rows) from a row-major LDS region.
// Lane layout (ISA 7.12.2): lanes 0-15 -> K {0..7,16..23}, lanes 16-31 -> +8.
// Unpredicated: rows >= MT read junk/zero, affecting only discarded D rows.
static __device__ __forceinline__ v16bf load_a(const u16* base, int rstride,
                                               int lane, int kbase) {
  int row = lane & 15;
  int off = row * rstride + kbase + ((lane & 16) ? 8 : 0);
  u32x4 lo = *(const u32x4*)(base + off);
  u32x4 hi = *(const u32x4*)(base + off + 16);
  return pack_ab(lo, hi);
}

// B operand (32x16 bf16) from a lane-order staged tile: 32 bytes/lane, contiguous.
static __device__ __forceinline__ v16bf load_b(const u16* tile, int lane) {
  const u32x4* p = (const u32x4*)(tile + lane * 16);
  return pack_ab(p[0], p[1]);
}

// ---- staging: global (lane-order bf16 tiles) -> LDS, ASYNCcnt-tracked -------
#if __has_builtin(__builtin_amdgcn_global_load_async_to_lds_b128) && \
    __has_builtin(__builtin_amdgcn_s_wait_asynccnt)
#define USE_ASYNC_LDS 1
#define ASYNC_WAIT(n) __builtin_amdgcn_s_wait_asynccnt(n)
#else
#define ASYNC_WAIT(n)
#endif

static __device__ __forceinline__ void stage_copy(u16* dst, const u16* src,
                                                  int nelem, int tid) {
  int n4 = nelem >> 3;
  const u32x4* s4 = (const u32x4*)src;
  u32x4* d4 = (u32x4*)dst;
#ifdef USE_ASYNC_LDS
  for (int i = tid; i < n4; i += NTHREADS)
    __builtin_amdgcn_global_load_async_to_lds_b128((v4i*)(s4 + i), (v4i*)(d4 + i), 0, 0);
#else
  for (int i = tid; i < n4; i += NTHREADS) d4[i] = s4[i];
#endif
}

// ---------------------------------------------------------------------------
// Prologue: repack fp32 weights -> bf16 WMMA lane-order tiles in ws.
// One thread per destination element.
// ---------------------------------------------------------------------------
__global__ __launch_bounds__(256) void wrepack_kernel(
    const float* __restrict__ w0, const float* __restrict__ w1,
    const float* __restrict__ w2, const float* __restrict__ w3,
    u16* __restrict__ ws) {
  int idx = blockIdx.x * 256 + threadIdx.x;
  if (idx >= WS_ELEMS) return;
  int region = idx >> 19;              // 512K elems per region
  int local  = idx & ((1 << 19) - 1);
  int e    = local & 15;               // element within lane
  int l    = (local >> 4) & 31;        // destination lane
  int rest = local >> 9;
  int colN = l & 15;
  int ksel = (l & 16) ? 8 : 0;         // B layout: lanes 16-31 hold K+8 (K {0..7,16..23} +8)
  float v;
  if (region == 0) {                   // w0: (64,64,128)
    int nt = rest & 7, kc = (rest >> 3) & 1, n = rest >> 4;
    int K = kc * 32 + ksel + (e & 7) + ((e & 8) << 1), h = nt * 16 + colN;
    v = w0[(n * 64 + K) * 128 + h];
  } else if (region == 1) {            // w1: (128,64,64)
    int nt = rest & 3, kc = (rest >> 2) & 1, s = rest >> 3;
    int K = kc * 32 + ksel + (e & 7) + ((e & 8) << 1), j = nt * 16 + colN;
    v = w1[(s * 64 + K) * 64 + j];
  } else if (region == 2) {            // w2: (128,64,64)
    int nt = rest & 3, kc = (rest >> 2) & 1, s = rest >> 3;
    int K = kc * 32 + ksel + (e & 7) + ((e & 8) << 1), j = nt * 16 + colN;
    v = w2[(s * 64 + K) * 64 + j];
  } else {                             // w3: (64,128,64)
    int nt = rest & 3, kc = (rest >> 2) & 3, t = rest >> 4;
    int K = kc * 32 + ksel + (e & 7) + ((e & 8) << 1), j = nt * 16 + colN;
    v = w3[(t * 128 + K) * 64 + j];
  }
  ws[idx] = f2bf(v);
}

// ---------------------------------------------------------------------------
// Fused kernel: one WG = 8 batch rows, all 4 stages, intermediates in LDS.
// Weight staging is double-buffered: copy(i+1) issued async, then
// s_wait_asynccnt(8) guarantees buffer i landed while the prefetch flies.
// ---------------------------------------------------------------------------
__global__ __launch_bounds__(NTHREADS) void monarch_fused_kernel(
    const float* __restrict__ x, const u16* __restrict__ ws,
    const float* __restrict__ b1, const float* __restrict__ b3,
    float* __restrict__ out) {
  __shared__ u16 y0buf[128 * 8 * 64];   // 128 KB  [s][b][k0]
  __shared__ u16 zx[64 * 8 * 128];      // 128 KB  xbuf[8][4096]  then  z[t][b][s]
  __shared__ u16 wstage[2][8192];       //  32 KB  double-buffered weight tiles
  __shared__ u16 elustage[8 * 64];      //   1 KB  elu(A) staging for stage 1_0

  const int tid  = threadIdx.x;
  const int lane = tid & 31;
  const int wave = tid >> 5;
  const int wg   = blockIdx.x;
  const int colN = lane & 15;
  const bool hiHalf = (lane & 16) != 0;        // lanes 16-31 hold rows 8..15

  // ---- prefetch first w0 block; stream x tile in (fp32 -> bf16, row-major) --
  stage_copy(wstage[0], ws + W0_OFF, 8192, tid);
  const float* xg = x + (size_t)wg * MT * IN_DIM;
  for (int i = tid * 2; i < MT * IN_DIM; i += NTHREADS * 2) {
    float a = xg[i], b = xg[i + 1];
    *(u32*)(zx + i) = f2bf2(a, b);      // paired convert + single b32 LDS store
  }

  // ---- phase A: stage 0_0 ---------------------------------------------------
  for (int n = 0; n < 64; ++n) {
    const u16* wcur = wstage[n & 1];
    if (n + 1 < 64) {
      stage_copy(wstage[(n + 1) & 1], ws + W0_OFF + (n + 1) * 8192, 8192, tid);
      ASYNC_WAIT(8);                    // buffer n complete; prefetch in flight
    } else {
      ASYNC_WAIT(0);
    }
    __syncthreads();
    for (int q = 0; q < 2; ++q) {
      int nt = wave * 2 + q;                    // 8 N-tiles over 4 waves
      v8f acc = {0.f, 0.f, 0.f, 0.f, 0.f, 0.f, 0.f, 0.f};
#pragma unroll
      for (int kc = 0; kc < 2; ++kc) {
        v16bf a = load_a(zx, IN_DIM, lane, n * 64 + kc * 32);
        v16bf b = load_b(wcur + (kc * 8 + nt) * 512, lane);
        acc = __builtin_amdgcn_wmma_f32_16x16x32_bf16(false, a, false, b,
                                                      (short)0, acc, false, false);
      }
      int s = nt * 16 + colN;
      if (!hiHalf) {
#pragma unroll
        for (int r = 0; r < 8; ++r)             // rows 0..7 live in lanes 0-15
          y0buf[(s * 8 + r) * 64 + n] = f2bf(acc[r]);
      }
    }
    __syncthreads();
  }

  // ---- phase B: stage 0_1 -> +b1 -> ELU -> stage 1_0 ------------------------
  stage_copy(wstage[0],        ws + W1_OFF, 4096, tid);
  stage_copy(wstage[0] + 4096, ws + W2_OFF, 4096, tid);
  for (int s = 0; s < 128; ++s) {
    const u16* wcur = wstage[s & 1];
    if (s + 1 < 128) {
      u16* nb = wstage[(s + 1) & 1];
      stage_copy(nb,        ws + W1_OFF + (s + 1) * 4096, 4096, tid);
      stage_copy(nb + 4096, ws + W2_OFF + (s + 1) * 4096, 4096, tid);
      ASYNC_WAIT(8);
    } else {
      ASYNC_WAIT(0);
    }
    __syncthreads();
    int nt = wave;                              // 4 N-tiles over 4 waves
    // stage 0_1: A_s[b,k0] = y0[s][b][k0]
    v8f acc = {0.f, 0.f, 0.f, 0.f, 0.f, 0.f, 0.f, 0.f};
#pragma unroll
    for (int kc = 0; kc < 2; ++kc) {
      v16bf a = load_a(y0buf + s * 8 * 64, 64, lane, kc * 32);
      v16bf b = load_b(wcur + (kc * 4 + nt) * 512, lane);
      acc = __builtin_amdgcn_wmma_f32_16x16x32_bf16(false, a, false, b,
                                                    (short)0, acc, false, false);
    }
    float bias = b1[s * 64 + nt * 16 + colN];
    if (!hiHalf) {
#pragma unroll
      for (int r = 0; r < 8; ++r) {
        float v = acc[r] + bias;
        v = (v > 0.f) ? v : (__expf(v) - 1.f);  // ELU
        elustage[r * 64 + nt * 16 + colN] = f2bf(v);
      }
    }
    __syncthreads();
    // stage 1_0: z_s[b,t] = elu(...) @ w2[s]
    v8f acc2 = {0.f, 0.f, 0.f, 0.f, 0.f, 0.f, 0.f, 0.f};
#pragma unroll
    for (int kc = 0; kc < 2; ++kc) {
      v16bf a = load_a(elustage, 64, lane, kc * 32);
      v16bf b = load_b(wcur + 4096 + (kc * 4 + nt) * 512, lane);
      acc2 = __builtin_amdgcn_wmma_f32_16x16x32_bf16(false, a, false, b,
                                                     (short)0, acc2, false, false);
    }
    int t = nt * 16 + colN;
    if (!hiHalf) {
#pragma unroll
      for (int r = 0; r < 8; ++r)
        zx[(t * 8 + r) * 128 + s] = f2bf(acc2[r]);   // z[t][b][s] (xbuf is dead)
    }
    __syncthreads();
  }

  // ---- phase C: stage 1_1 ---------------------------------------------------
  stage_copy(wstage[0], ws + W3_OFF, 8192, tid);
  for (int t = 0; t < 64; ++t) {
    const u16* wcur = wstage[t & 1];
    if (t + 1 < 64) {
      stage_copy(wstage[(t + 1) & 1], ws + W3_OFF + (t + 1) * 8192, 8192, tid);
      ASYNC_WAIT(8);
    } else {
      ASYNC_WAIT(0);
    }
    __syncthreads();
    int nt = wave;
    v8f acc = {0.f, 0.f, 0.f, 0.f, 0.f, 0.f, 0.f, 0.f};
#pragma unroll
    for (int kc = 0; kc < 4; ++kc) {
      v16bf a = load_a(zx + t * 8 * 128, 128, lane, kc * 32);
      v16bf b = load_b(wcur + (kc * 4 + nt) * 512, lane);
      acc = __builtin_amdgcn_wmma_f32_16x16x32_bf16(false, a, false, b,
                                                    (short)0, acc, false, false);
    }
    int col = t * 64 + nt * 16 + colN;
    float bias = b3[t * 64 + nt * 16 + colN];
    if (!hiHalf) {
#pragma unroll
      for (int r = 0; r < 8; ++r)
        out[(size_t)(wg * MT + r) * IN_DIM + col] = acc[r] + bias;
    }
    __syncthreads();
  }
}

// ---------------------------------------------------------------------------
extern "C" void kernel_launch(void* const* d_in, const int* in_sizes, int n_in,
                              void* d_out, int out_size, void* d_ws, size_t ws_size,
                              hipStream_t stream) {
  const float* x  = (const float*)d_in[0];
  const float* w0 = (const float*)d_in[1];
  const float* w1 = (const float*)d_in[2];
  const float* b1 = (const float*)d_in[3];
  const float* w2 = (const float*)d_in[4];
  const float* w3 = (const float*)d_in[5];
  const float* b3 = (const float*)d_in[6];
  u16* ws = (u16*)d_ws;                 // needs 4 MB of scratch

  wrepack_kernel<<<WS_ELEMS / 256, 256, 0, stream>>>(w0, w1, w2, w3, ws);
  monarch_fused_kernel<<<NROWS / MT, NTHREADS, 0, stream>>>(x, ws, b1, b3,
                                                            (float*)d_out);
}
